// GraphEmbedder_64707977281671
// MI455X (gfx1250) — matrix operand order repaired and checked by reading, not compiled, not run on hardware
//
#include <hip/hip_runtime.h>

typedef __attribute__((ext_vector_type(16))) __bf16 v16bf;
typedef __attribute__((ext_vector_type(8)))  float  v8f;
typedef __attribute__((ext_vector_type(4)))  unsigned int u32x4;

#define Ek   256   // edges per graph
#define Nk   128   // max nodes
#define Fk   64    // node feature width
#define Dk   256   // output dim
#define PREk 192   // N + F
#define AS   136   // padded LDS row stride in bf16 elems (spreads banks)

__global__ __launch_bounds__(256, 1)
void graph_embed_kernel(const int* __restrict__ edge_index,
                        const float* __restrict__ edge_weight,
                        const int* __restrict__ node_feat_idx,
                        const float* __restrict__ W,
                        const float* __restrict__ bias,
                        float* __restrict__ out)
{
    // Adjacency block (bf16), 128x136 -> 34.8 KB
    __shared__ __align__(16) __bf16 sA[Nk * AS];
    // W[0:128,:]^T as bf16 [n][k], 256x136 -> 69.6 KB
    __shared__ __align__(16) __bf16 sWt[Dk * AS];
    __shared__ int sNfi[Nk];

    const int b    = blockIdx.x;
    const int tid  = threadIdx.x;
    const int lane = tid & 31;        // wave32
    const int wave = tid >> 5;        // 0..7

    // ---- zero adjacency tile (dword stores) ----
    {
        unsigned int* z = (unsigned int*)sA;
        const int n32 = (Nk * AS) / 2;
        for (int i = tid; i < n32; i += 256) z[i] = 0u;
    }

    // ---- stage Wt = W[0:128,:]^T as bf16 (coalesced global reads, L2-resident W) ----
    for (int k = 0; k < Nk; ++k) {
        float w = W[k * Dk + tid];            // row k, col = tid
        sWt[tid * AS + k] = (__bf16)w;
    }

    // ---- node feature indices ----
    if (tid < Nk) sNfi[tid] = node_feat_idx[b * Nk + tid];

    __syncthreads();

    // ---- scatter edges (set semantics; symmetric); inputs touched once -> NT loads ----
    {
        const int e = tid;                    // E == 256 == blockDim.x
        int i1 = __builtin_nontemporal_load(&edge_index[(b * Ek + e) * 2 + 0]);
        int i2 = __builtin_nontemporal_load(&edge_index[(b * Ek + e) * 2 + 1]);
        __bf16 w = (__bf16)__builtin_nontemporal_load(&edge_weight[b * Ek + e]);
        sA[i1 * AS + i2] = w;
        sA[i2 * AS + i1] = w;
    }

    __syncthreads();

    // ---- per-wave GEMM: C[16,256] = A[16,128] @ W[0:128, 0:256] ----
    const int mtile   = wave;                 // 8 waves x 16 rows = 128 rows
    const int halfsel = lane >> 4;            // 0: lanes 0-15, 1: lanes 16-31
    const int l16     = lane & 15;

    // Preload the 4 A-fragments for this row tile (ISA 16-bit A 16x32 layout:
    // lanes 0-15 hold K {0..7,16..23}, lanes 16-31 hold K {8..15,24..31})
    v16bf afrag[4];
    {
        const int row = mtile * 16 + l16;
        #pragma unroll
        for (int kt = 0; kt < 4; ++kt) {
            const int k0 = kt * 32 + halfsel * 8;
            union { v16bf v; u32x4 q[2]; } fa;
            fa.q[0] = *(const u32x4*)&sA[row * AS + k0];
            fa.q[1] = *(const u32x4*)&sA[row * AS + k0 + 16];
            afrag[kt] = fa.v;
        }
    }

    // One-hot row base pointers for the 8 C-VGPR rows this lane covers
    const float* wrow[8];
    #pragma unroll
    for (int r = 0; r < 8; ++r) {
        int m = mtile * 16 + r + halfsel * 8;
        wrow[r] = W + (size_t)(Nk + sNfi[m]) * Dk;
    }

    float* outb = out + (size_t)b * Nk * Dk;

    for (int nt = 0; nt < 16; ++nt) {
        const int col = nt * 16 + l16;

        // B fragment (ISA 16-bit B 32x16 layout: lane = column, lanes 0-15 hold K {0..7,16..23}, lanes 16-31 hold K {8..15,24..31})
        v8f acc = {};
        #pragma unroll
        for (int kt = 0; kt < 4; ++kt) {
            const int k0 = kt * 32 + halfsel * 8;
            union { v16bf v; u32x4 q[2]; } fb;
            fb.q[0] = *(const u32x4*)&sWt[col * AS + k0];
            fb.q[1] = *(const u32x4*)&sWt[col * AS + k0 + 16];
            acc = __builtin_amdgcn_wmma_f32_16x16x32_bf16(
                false, afrag[kt], false, fb.v, (short)0, acc, false, false);
        }

        // Exact f32 epilogue: bias + one-hot gather (L2-resident W, regular TH),
        // then stream the output with non-temporal stores (don't pollute L2).
        const float bc = bias[col];
        #pragma unroll
        for (int r = 0; r < 8; ++r) {
            int m = mtile * 16 + r + halfsel * 8;
            float v = acc[r] + bc + wrow[r][col];
            __builtin_nontemporal_store(v, &outb[m * Dk + col]);
        }
    }
}

extern "C" void kernel_launch(void* const* d_in, const int* in_sizes, int n_in,
                              void* d_out, int out_size, void* d_ws, size_t ws_size,
                              hipStream_t stream) {
    const int*   edge_index    = (const int*)d_in[0];
    const float* edge_weight   = (const float*)d_in[1];
    const int*   node_feat_idx = (const int*)d_in[2];
    const float* W             = (const float*)d_in[3];
    const float* bias          = (const float*)d_in[4];
    float*       out           = (float*)d_out;

    const int Bsz = in_sizes[1] / Ek;   // edge_weight is [B, E]
    graph_embed_kernel<<<Bsz, 256, 0, stream>>>(edge_index, edge_weight,
                                                node_feat_idx, W, bias, out);
}
